// AttentionModelTF_50070728737101
// MI455X (gfx1250) — compile-verified, matrix operand order checked by reading, not run on hardware
//
#include <hip/hip_runtime.h>
#include <hip/hip_bf16.h>
#include <math.h>

typedef _Float16 h16;
typedef __attribute__((ext_vector_type(16))) _Float16 v16h;
typedef __attribute__((ext_vector_type(8)))  _Float16 v8h;
typedef __attribute__((ext_vector_type(8)))  float    v8f;
typedef __attribute__((ext_vector_type(4)))  unsigned u32x4;
typedef __attribute__((ext_vector_type(8)))  int      i32x8;
typedef __attribute__((ext_vector_type(4)))  int      i32x4;

static constexpr int Bn   = 1024;   // batch
static constexpr int Nn   = 100;    // nodes
static constexpr int TOK  = Bn * Nn;
static constexpr int Dd   = 128;
static constexpr int Hh   = 8;
static constexpr int DHd  = 16;
static constexpr int Ff   = 512;
static constexpr int Ll   = 3;

#if __has_include(<hip/amd_detail/amd_gfx1250_TDM.h>)
#define TDM_SIX_ARG 1
#endif

// ---------------------------------------------------------------------------
// TDM: issue one 2D tensor_load_to_lds (wave-level op; EXEC ignored).
// D# bitfields per CDNA5 ISA ch.8: group0 = {count|flags, lds_addr,
// global_addr[31:0], global_addr[56:32]|type=2}; group1 packs data_size,
// tensor_dim0/1, tile_dim0/1/2, tensor_dim0_stride. data_size = 8B units.
// ---------------------------------------------------------------------------
__device__ static inline void tdm_load_2d(const void* gptr, unsigned lds_off,
                                          unsigned rowUnits /* row len, 8B units */,
                                          unsigned tileRows, unsigned tensorRows) {
  unsigned long long ga = (unsigned long long)(size_t)gptr;
  u32x4 g0;
  g0[0] = 1u;                                        // count=1, user descriptor
  g0[1] = lds_off;                                   // lds_addr (bytes)
  g0[2] = (unsigned)(ga & 0xFFFFFFFFu);              // global_addr[31:0]
  g0[3] = (unsigned)((ga >> 32) & 0x01FFFFFFu)       // global_addr[56:32]
        | (2u << 30);                                // type = 2 ("image")
  unsigned long long st0 = rowUnits;                 // tensor_dim0_stride (units)
  i32x8 g1;
  g1[0] = (int)(3u << 16);                                            // data_size=8B, wgmask=0
  g1[1] = (int)((rowUnits & 0xFFFFu) << 16);                          // tensor_dim0 lo16
  g1[2] = (int)(((rowUnits >> 16) & 0xFFFFu) | ((tensorRows & 0xFFFFu) << 16)); // td0 hi | td1 lo
  g1[3] = (int)(((tensorRows >> 16) & 0xFFFFu) | ((rowUnits & 0xFFFFu) << 16)); // td1 hi | tile_dim0
  g1[4] = (int)(tileRows & 0xFFFFu);                                  // tile_dim1 (tile_dim2=0)
  g1[5] = (int)(st0 & 0xFFFFFFFFull);                                 // stride0 lo32
  g1[6] = (int)((st0 >> 32) & 0xFFFFull);                             // stride0 hi16 | stride1 lo16
  g1[7] = 0;                                                          // stride1 hi
  i32x4 z4 = {0, 0, 0, 0};
#ifdef TDM_SIX_ARG
  i32x8 z8 = {0, 0, 0, 0, 0, 0, 0, 0};
  __builtin_amdgcn_tensor_load_to_lds(g0, g1, z4, z4, z8, 0);
#else
  __builtin_amdgcn_tensor_load_to_lds(g0, g1, z4, z4, 0);
#endif
}

// ---------------------------------------------------------------------------
// Weight prep: f32 -> f16, transposed to (N_out, K_in) row-major so WMMA B
// fragments load contiguously (lane = N, contiguous K), matching A loads.
// ---------------------------------------------------------------------------
__global__ void prep_qkv_w(const float* __restrict__ Wq, const float* __restrict__ Wk,
                           const float* __restrict__ Wv, const float* __restrict__ bq,
                           const float* __restrict__ bk, const float* __restrict__ bv,
                           h16* __restrict__ WqkvT, float* __restrict__ bqkv) {
  int t = blockIdx.x * blockDim.x + threadIdx.x;          // Ll*384*128
  if (t >= Ll * 384 * Dd) return;
  int k = t & 127; int rest = t >> 7;
  int n = rest % 384; int l = rest / 384;
  const float* W; int col;
  if (n < 128)      { W = Wq; col = n; }
  else if (n < 256) { W = Wk; col = n - 128; }
  else              { W = Wv; col = n - 256; }
  WqkvT[t] = (h16)W[((size_t)l * Dd + k) * Dd + col];
  if (k == 0) {
    const float* bb = (n < 128) ? bq : (n < 256 ? bk : bv);
    bqkv[l * 384 + n] = bb[l * Dd + col];
  }
}

__global__ void prep_wo(const float* __restrict__ Wo, h16* __restrict__ WoT) {
  int t = blockIdx.x * blockDim.x + threadIdx.x;          // Ll*128*128
  if (t >= Ll * Dd * Dd) return;
  int k = t & 127; int rest = t >> 7;
  int dout = rest % Dd; int l = rest / Dd;
  WoT[t] = (h16)Wo[((size_t)l * Dd + k) * Dd + dout];     // Wo (L,128_in,128_out)
}

__global__ void prep_w1(const float* __restrict__ W1, h16* __restrict__ W1T) {
  int t = blockIdx.x * blockDim.x + threadIdx.x;          // Ll*512*128
  if (t >= Ll * Ff * Dd) return;
  int k = t & 127; int rest = t >> 7;
  int f = rest % Ff; int l = rest / Ff;
  W1T[t] = (h16)W1[((size_t)l * Dd + k) * Ff + f];
}

__global__ void prep_w2(const float* __restrict__ W2, h16* __restrict__ W2T) {
  int t = blockIdx.x * blockDim.x + threadIdx.x;          // Ll*128*512
  if (t >= Ll * Dd * Ff) return;
  int k = t % Ff; int rest = t / Ff;
  int dout = rest % Dd; int l = rest / Dd;
  W2T[t] = (h16)W2[((size_t)l * Ff + k) * Dd + dout];
}

__global__ void prep_dec(const float* __restrict__ dec_w, h16* __restrict__ decT) {
  int t = blockIdx.x * blockDim.x + threadIdx.x;          // 128*384
  if (t >= Dd * 384) return;
  int k = t % 384; int n = t / 384;
  decT[t] = (h16)dec_w[(size_t)k * Dd + n];               // dec_w (384,128)
}

// ---------------------------------------------------------------------------
// Embedding: x = coords @ emb_w + emb_b   (K=2, trivial on VALU)
// ---------------------------------------------------------------------------
__global__ void embed_kernel(const float* __restrict__ coords, const float* __restrict__ emb_w,
                             const float* __restrict__ emb_b,
                             float* __restrict__ x, h16* __restrict__ xh) {
  int t = blockIdx.x * blockDim.x + threadIdx.x;          // TOK*128
  if (t >= TOK * Dd) return;
  int d = t & 127; size_t tok = (size_t)(t >> 7);
  float c0 = coords[tok * 2 + 0], c1 = coords[tok * 2 + 1];
  float v = fmaf(c0, emb_w[d], fmaf(c1, emb_w[Dd + d], emb_b[d]));
  x[t] = v; xh[t] = (h16)v;
}

// ---------------------------------------------------------------------------
// Register-blocked WMMA GEMM for the big encoder GEMMs.
// Block = 8 waves; block computes a 512(M) x 64(N) strip; each wave owns
// 64x64 (4x4 tiles of 16x16, 16 v8f accumulators -> ~32 FLOP/byte).
// B strip (64 rows x K halves, <=64KB) is staged into LDS once per block by
// the Tensor Data Mover; fragments then come from ds_load_b128.
// Requires M%512==0, N%64==0, K%32==0.
// ---------------------------------------------------------------------------
__global__ __launch_bounds__(256) void wmma_gemm_blk(
    const h16* __restrict__ A, const h16* __restrict__ BT,
    const float* __restrict__ bias, float* __restrict__ Cf, h16* __restrict__ Ch,
    int M, int N, int K, int relu) {
  __shared__ __align__(16) h16 lds_b[64 * 512];
  int stripsM = M >> 9;
  int bm = blockIdx.x % stripsM;
  int bn = blockIdx.x / stripsM;
  int n0 = bn * 64;
  int wave = threadIdx.x >> 5;
  int lane = threadIdx.x & 31;
  int hsel = (lane >> 4) & 1;
  int rsel = lane & 15;

  if (wave == 0) {                                       // wave-level TDM issue
    tdm_load_2d(BT + (size_t)n0 * K, (unsigned)(size_t)(void*)lds_b,
                (unsigned)(K >> 2), 64u, (unsigned)(N - n0));
    __builtin_amdgcn_s_wait_tensorcnt(0);
  }
  __syncthreads();

  int m0 = bm * 512 + wave * 64;
  const h16* arow0 = A + (size_t)(m0 + rsel) * K + hsel * 8;

  v8f zero = {};
  v8f acc[4][4];
#pragma unroll
  for (int i = 0; i < 4; ++i)
#pragma unroll
    for (int j = 0; j < 4; ++j) acc[i][j] = zero;

  for (int k0 = 0; k0 < K; k0 += 32) {
    v16h a[4], b[4];
#pragma unroll
    for (int i = 0; i < 4; ++i) {
      const h16* ap = arow0 + (size_t)i * 16 * K + k0;
      __builtin_prefetch((const void*)(ap + 64), 0, 0);
      v8h lo = *(const v8h*)(ap);
      v8h hi = *(const v8h*)(ap + 16);
#pragma unroll
      for (int e = 0; e < 8; ++e) { a[i][e] = lo[e]; a[i][e + 8] = hi[e]; }
    }
#pragma unroll
    for (int j = 0; j < 4; ++j) {
      const h16* bp = lds_b + (size_t)(j * 16 + rsel) * K + k0 + hsel * 8;
      v8h lo = *(const v8h*)(bp);
      v8h hi = *(const v8h*)(bp + 16);
#pragma unroll
      for (int e = 0; e < 8; ++e) { b[j][e] = lo[e]; b[j][e + 8] = hi[e]; }
    }
#pragma unroll
    for (int i = 0; i < 4; ++i)
#pragma unroll
      for (int j = 0; j < 4; ++j)
        acc[i][j] = __builtin_amdgcn_wmma_f32_16x16x32_f16(
            false, a[i], false, b[j], (short)0, acc[i][j], false, false);
  }

#pragma unroll
  for (int j = 0; j < 4; ++j) {
    int n = n0 + j * 16 + rsel;
    float bc = bias ? bias[n] : 0.0f;
#pragma unroll
    for (int i = 0; i < 4; ++i) {
      int mb = m0 + i * 16 + (hsel ? 8 : 0);
#pragma unroll
      for (int r = 0; r < 8; ++r) {
        float v = acc[i][j][r] + bc;
        if (relu) v = fmaxf(v, 0.0f);
        size_t idx = (size_t)(mb + r) * N + n;
        if (Cf) Cf[idx] = v;
        if (Ch) Ch[idx] = (h16)v;
      }
    }
  }
}

// ---------------------------------------------------------------------------
// One-tile-per-wave WMMA GEMM (kept for the small decoder GEMM: M=1024,
// where fine-grained tiles give 512 waves instead of 4 blocks).
// ---------------------------------------------------------------------------
__global__ __launch_bounds__(256) void wmma_gemm(
    const h16* __restrict__ A, const h16* __restrict__ BT,
    const float* __restrict__ bias, float* __restrict__ Cf, h16* __restrict__ Ch,
    int M, int N, int K, int relu) {
  int tilesM = M >> 4, tilesN = N >> 4;
  int wid = blockIdx.x * (blockDim.x >> 5) + (threadIdx.x >> 5);
  if (wid >= tilesM * tilesN) return;                     // wave-uniform exit
  int tm = wid % tilesM, tn = wid / tilesM;
  int lane = threadIdx.x & 31;
  int hsel = (lane >> 4) & 1;
  int rsel = lane & 15;

  const h16* arow = A  + (size_t)(tm * 16 + rsel) * K + hsel * 8;
  const h16* brow = BT + (size_t)(tn * 16 + rsel) * K + hsel * 8;

  v8f acc = {};
  for (int k0 = 0; k0 < K; k0 += 32) {
    v8h a0 = *(const v8h*)(arow + k0);
    v8h a1 = *(const v8h*)(arow + k0 + 16);
    v8h b0 = *(const v8h*)(brow + k0);
    v8h b1 = *(const v8h*)(brow + k0 + 16);
    v16h af, bf;
#pragma unroll
    for (int i = 0; i < 8; ++i) {
      af[i] = a0[i]; af[i + 8] = a1[i];
      bf[i] = b0[i]; bf[i + 8] = b1[i];
    }
    acc = __builtin_amdgcn_wmma_f32_16x16x32_f16(
        false, af, false, bf, (short)0, acc, false, false);
  }
  int n = tn * 16 + rsel;
  float bc = bias ? bias[n] : 0.0f;
  int mbase = tm * 16 + (hsel ? 8 : 0);
#pragma unroll
  for (int i = 0; i < 8; ++i) {
    float v = acc[i] + bc;
    if (relu) v = fmaxf(v, 0.0f);
    size_t idx = (size_t)(mbase + i) * N + n;
    if (Cf) Cf[idx] = v;
    if (Ch) Ch[idx] = (h16)v;
  }
}

// ---------------------------------------------------------------------------
// Attention: online-softmax, one thread per (b,h,q). DH=16, N=100.
// qkv layout per token: [q(128) | k(128) | v(128)] f16.
// ---------------------------------------------------------------------------
__global__ void attn_kernel(const h16* __restrict__ qkv, h16* __restrict__ o) {
  int t = blockIdx.x * blockDim.x + threadIdx.x;
  if (t >= Bn * Hh * Nn) return;
  int qi = t % Nn; int bh = t / Nn;
  int h = bh % Hh;  int b = bh / Hh;

  const h16* qrow = qkv + (size_t)(b * Nn + qi) * 384 + h * DHd;
  float qv[DHd];
#pragma unroll
  for (int d = 0; d < DHd; ++d) qv[d] = (float)qrow[d];

  const h16* kbase = qkv + (size_t)b * Nn * 384 + 128 + h * DHd;
  const h16* vbase = qkv + (size_t)b * Nn * 384 + 256 + h * DHd;

  float m = -INFINITY, s = 0.0f, acc[DHd];
#pragma unroll
  for (int d = 0; d < DHd; ++d) acc[d] = 0.0f;

  for (int n = 0; n < Nn; ++n) {
    const h16* kr = kbase + (size_t)n * 384;
    float dot = 0.0f;
#pragma unroll
    for (int d = 0; d < DHd; ++d) dot = fmaf(qv[d], (float)kr[d], dot);
    dot *= 0.25f;                                         // 1/sqrt(DH)
    float mn = fmaxf(m, dot);
    float corr = __expf(m - mn);
    float p = __expf(dot - mn);
    s = s * corr + p;
    const h16* vr = vbase + (size_t)n * 384;
#pragma unroll
    for (int d = 0; d < DHd; ++d) acc[d] = fmaf(acc[d], corr, p * (float)vr[d]);
    m = mn;
  }
  float inv = 1.0f / s;
  h16* orow = o + (size_t)(b * Nn + qi) * Dd + h * DHd;
#pragma unroll
  for (int d = 0; d < DHd; ++d) orow[d] = (h16)(acc[d] * inv);
}

// ---------------------------------------------------------------------------
// Residual + LayerNorm over D=128. One wave32 per token, 4 elems/lane.
// ---------------------------------------------------------------------------
__global__ void ln_add_kernel(float* __restrict__ x, h16* __restrict__ xh,
                              const float* __restrict__ delta,
                              const float* __restrict__ g, const float* __restrict__ b) {
  int wid  = (blockIdx.x * blockDim.x + threadIdx.x) >> 5;
  int lane = threadIdx.x & 31;
  if (wid >= TOK) return;
  size_t base = (size_t)wid * Dd;
  float v[4], sum = 0.0f;
#pragma unroll
  for (int i = 0; i < 4; ++i) {
    int d = lane + i * 32;
    v[i] = x[base + d] + delta[base + d];
    sum += v[i];
  }
#pragma unroll
  for (int off = 16; off > 0; off >>= 1) sum += __shfl_xor(sum, off, 32);
  float mean = sum * (1.0f / 128.0f);
  float var = 0.0f;
#pragma unroll
  for (int i = 0; i < 4; ++i) { float dv = v[i] - mean; var = fmaf(dv, dv, var); }
#pragma unroll
  for (int off = 16; off > 0; off >>= 1) var += __shfl_xor(var, off, 32);
  var *= (1.0f / 128.0f);
  float r = rsqrtf(var + 1e-6f);
#pragma unroll
  for (int i = 0; i < 4; ++i) {
    int d = lane + i * 32;
    float y = fmaf((v[i] - mean) * r, g[d], b[d]);
    x[base + d] = y; xh[base + d] = (h16)y;
  }
}

// ---------------------------------------------------------------------------
// Decoder helpers
// ---------------------------------------------------------------------------
__global__ void pool_kernel(const float* __restrict__ enc,
                            float* __restrict__ pooled, float* __restrict__ first) {
  int t = blockIdx.x * blockDim.x + threadIdx.x;          // Bn*128
  if (t >= Bn * Dd) return;
  int d = t & 127; int b = t >> 7;
  const float* base = enc + (size_t)b * Nn * Dd + d;
  float s = 0.0f;
  for (int n = 0; n < Nn; ++n) s += base[(size_t)n * Dd];
  pooled[t] = s * (1.0f / (float)Nn);
  first[t]  = base[0];
}

__global__ void dec_init(int* __restrict__ mask, int* __restrict__ cur) {
  int t = blockIdx.x * blockDim.x + threadIdx.x;
  if (t < Bn * Nn) mask[t] = 1;
  if (t < Bn) cur[t] = 0;
}

// S1: record tour[step]=cur, mask it out, build ctx = [pooled|enc[cur]|first] f16
__global__ void dec_step_build(const float* __restrict__ pooled, const float* __restrict__ first,
                               const float* __restrict__ enc, const int* __restrict__ cur,
                               int* __restrict__ mask, int* __restrict__ tour_out,
                               h16* __restrict__ ctx, int step) {
  int b = blockIdx.x;
  int t = threadIdx.x;                                    // 384 threads
  int c = cur[b];
  if (t == 0) {
    tour_out[b * Nn + step] = c;
    mask[b * Nn + c] = 0;
  }
  float v;
  if (t < 128)      v = pooled[b * Dd + t];
  else if (t < 256) v = enc[((size_t)b * Nn + c) * Dd + (t - 128)];
  else              v = first[b * Dd + (t - 256)];
  ctx[(size_t)b * 384 + t] = (h16)v;
}

// S3: scores = CLIP*tanh(q.enc/sqrt(D)); masked argmax (first-index tiebreak)
__global__ void dec_step_score(const float* __restrict__ qbuf, const float* __restrict__ enc,
                               const int* __restrict__ mask, int* __restrict__ cur) {
  __shared__ float qs[Dd];
  __shared__ float sv[128];
  __shared__ int   si[128];
  int b = blockIdx.x, t = threadIdx.x;                    // 128 threads
  qs[t] = qbuf[b * Dd + t];
  __syncthreads();
  float val = -INFINITY;
  if (t < Nn && mask[b * Nn + t]) {
    const float* er = enc + ((size_t)b * Nn + t) * Dd;
    float dot = 0.0f;
#pragma unroll 4
    for (int d = 0; d < Dd; ++d) dot = fmaf(qs[d], er[d], dot);
    val = 10.0f * tanhf(dot * 0.08838834764831845f);      // CLIP * tanh(s / sqrt(128))
  }
  sv[t] = val; si[t] = t;
  __syncthreads();
  for (int off = 64; off > 0; off >>= 1) {
    if (t < off) {
      bool take = (sv[t + off] > sv[t]) || (sv[t + off] == sv[t] && si[t + off] < si[t]);
      if (take) { sv[t] = sv[t + off]; si[t] = si[t + off]; }
    }
    __syncthreads();
  }
  if (t == 0) cur[b] = si[0];
}

__global__ void dec_final(const int* __restrict__ cur, int* __restrict__ tour_out) {
  int b = blockIdx.x * blockDim.x + threadIdx.x;
  if (b < Bn) tour_out[b * Nn + (Nn - 1)] = cur[b];
}

__global__ void tour_length(const float* __restrict__ coords, const int* __restrict__ tour_out,
                            float* __restrict__ lengths) {
  int b = blockIdx.x * blockDim.x + threadIdx.x;
  if (b >= Bn) return;
  float s = 0.0f;
  for (int i = 0; i < Nn; ++i) {
    int a = tour_out[b * Nn + i];
    int c = tour_out[b * Nn + ((i + 1) % Nn)];
    float dx = coords[((size_t)b * Nn + a) * 2 + 0] - coords[((size_t)b * Nn + c) * 2 + 0];
    float dy = coords[((size_t)b * Nn + a) * 2 + 1] - coords[((size_t)b * Nn + c) * 2 + 1];
    s += sqrtf(fmaf(dx, dx, dy * dy));
  }
  lengths[b] = s;
}

__global__ void copy_enc(const float* __restrict__ enc, float* __restrict__ out) {
  int t = blockIdx.x * blockDim.x + threadIdx.x;
  if (t < TOK * Dd) out[t] = enc[t];
}

// ---------------------------------------------------------------------------
// Host orchestration
// ---------------------------------------------------------------------------
extern "C" void kernel_launch(void* const* d_in, const int* in_sizes, int n_in,
                              void* d_out, int out_size, void* d_ws, size_t ws_size,
                              hipStream_t stream) {
  const float* coords = (const float*)d_in[0];
  const float* emb_w  = (const float*)d_in[1];
  const float* emb_b  = (const float*)d_in[2];
  const float* Wq     = (const float*)d_in[3];
  const float* bq     = (const float*)d_in[4];
  const float* Wk     = (const float*)d_in[5];
  const float* bk     = (const float*)d_in[6];
  const float* Wv     = (const float*)d_in[7];
  const float* bv     = (const float*)d_in[8];
  const float* Wo     = (const float*)d_in[9];
  const float* bo     = (const float*)d_in[10];
  const float* ln1_g  = (const float*)d_in[11];
  const float* ln1_b  = (const float*)d_in[12];
  const float* W1     = (const float*)d_in[13];
  const float* b1     = (const float*)d_in[14];
  const float* W2     = (const float*)d_in[15];
  const float* b2     = (const float*)d_in[16];
  const float* ln2_g  = (const float*)d_in[17];
  const float* ln2_b  = (const float*)d_in[18];
  const float* dec_w  = (const float*)d_in[19];
  const float* dec_b  = (const float*)d_in[20];

  // Workspace arena
  char* ws = (char*)d_ws;
  size_t off = 0;
  auto alloc = [&](size_t bytes) -> void* {
    void* p = ws + off;
    off = (off + bytes + 255) & ~(size_t)255;
    return p;
  };
  float* x_f32  = (float*)alloc((size_t)TOK * Dd * 4);    // activations (becomes enc)
  h16*   x_h    = (h16*)  alloc((size_t)TOK * Dd * 2);    // f16 mirror for WMMA A
  h16*   big_h  = (h16*)  alloc((size_t)TOK * Ff * 2);    // qkv (TOK*384) / ffn hidden
  h16*   o_h    = (h16*)  alloc((size_t)TOK * Dd * 2);    // attention output heads
  float* tmp32  = (float*)alloc((size_t)TOK * Dd * 4);    // proj / ffn output before LN
  h16*   WqkvT  = (h16*)  alloc((size_t)Ll * 384 * Dd * 2);
  float* bqkv   = (float*)alloc((size_t)Ll * 384 * 4);
  h16*   WoT    = (h16*)  alloc((size_t)Ll * Dd * Dd * 2);
  h16*   W1T    = (h16*)  alloc((size_t)Ll * Ff * Dd * 2);
  h16*   W2T    = (h16*)  alloc((size_t)Ll * Dd * Ff * 2);
  h16*   decT   = (h16*)  alloc((size_t)Dd * 384 * 2);
  float* pooled = (float*)alloc((size_t)Bn * Dd * 4);
  float* first  = (float*)alloc((size_t)Bn * Dd * 4);
  h16*   ctx_h  = (h16*)  alloc((size_t)Bn * 384 * 2);
  float* qbuf   = (float*)alloc((size_t)Bn * Dd * 4);
  int*   mask   = (int*)  alloc((size_t)Bn * Nn * 4);
  int*   cur    = (int*)  alloc((size_t)Bn * 4);

  // Output regions: [tour int32 bits | lengths f32 | enc f32]
  int*   tour_out = (int*)d_out;
  float* len_out  = (float*)d_out + (size_t)Bn * Nn;
  float* enc_out  = (float*)d_out + (size_t)Bn * Nn + Bn;

  auto cdiv = [](long long a, long long b) { return (int)((a + b - 1) / b); };
  // Encoder GEMMs: register-blocked + TDM-staged B (M%512==0, N%64==0)
  auto gemm_big = [&](const h16* A, const h16* BT, const float* bias,
                      float* Cf, h16* Ch, int M, int N, int K, int relu) {
    int blocks = (M >> 9) * (N >> 6);
    wmma_gemm_blk<<<blocks, 256, 0, stream>>>(A, BT, bias, Cf, Ch, M, N, K, relu);
  };
  // Small decoder GEMM: one 16x16 tile per wave
  auto gemm_small = [&](const h16* A, const h16* BT, const float* bias,
                        float* Cf, h16* Ch, int M, int N, int K, int relu) {
    int tiles = (M >> 4) * (N >> 4);
    wmma_gemm<<<cdiv(tiles, 8), 256, 0, stream>>>(A, BT, bias, Cf, Ch, M, N, K, relu);
  };

  // ---- weight prep (every call; deterministic) ----
  prep_qkv_w<<<cdiv((size_t)Ll * 384 * Dd, 256), 256, 0, stream>>>(Wq, Wk, Wv, bq, bk, bv, WqkvT, bqkv);
  prep_wo  <<<cdiv((size_t)Ll * Dd * Dd, 256), 256, 0, stream>>>(Wo, WoT);
  prep_w1  <<<cdiv((size_t)Ll * Ff * Dd, 256), 256, 0, stream>>>(W1, W1T);
  prep_w2  <<<cdiv((size_t)Ll * Dd * Ff, 256), 256, 0, stream>>>(W2, W2T);
  prep_dec <<<cdiv((size_t)Dd * 384, 256), 256, 0, stream>>>(dec_w, decT);

  // ---- encoder ----
  embed_kernel<<<cdiv((size_t)TOK * Dd, 256), 256, 0, stream>>>(coords, emb_w, emb_b, x_f32, x_h);

  for (int l = 0; l < Ll; ++l) {
    // fused QKV projection: (TOK x 128) x (128 x 384)
    gemm_big(x_h, WqkvT + (size_t)l * 384 * Dd, bqkv + l * 384, nullptr, big_h, TOK, 384, Dd, 0);
    // attention
    attn_kernel<<<cdiv((size_t)Bn * Hh * Nn, 128), 128, 0, stream>>>(big_h, o_h);
    // output projection
    gemm_big(o_h, WoT + (size_t)l * Dd * Dd, bo + l * Dd, tmp32, nullptr, TOK, Dd, Dd, 0);
    // x = LN(x + attn_out)
    ln_add_kernel<<<cdiv((size_t)TOK * 32, 256), 256, 0, stream>>>(x_f32, x_h, tmp32,
                                                                   ln1_g + l * Dd, ln1_b + l * Dd);
    // FFN
    gemm_big(x_h, W1T + (size_t)l * Ff * Dd, b1 + l * Ff, nullptr, big_h, TOK, Ff, Dd, 1);
    gemm_big(big_h, W2T + (size_t)l * Dd * Ff, b2 + l * Dd, tmp32, nullptr, TOK, Dd, Ff, 0);
    // x = LN(x + ffn_out)
    ln_add_kernel<<<cdiv((size_t)TOK * 32, 256), 256, 0, stream>>>(x_f32, x_h, tmp32,
                                                                   ln2_g + l * Dd, ln2_b + l * Dd);
  }

  // ---- decoder ----
  pool_kernel<<<cdiv((size_t)Bn * Dd, 256), 256, 0, stream>>>(x_f32, pooled, first);
  dec_init<<<cdiv((size_t)Bn * Nn, 256), 256, 0, stream>>>(mask, cur);

  for (int step = 0; step < Nn - 1; ++step) {
    dec_step_build<<<Bn, 384, 0, stream>>>(pooled, first, x_f32, cur, mask, tour_out, ctx_h, step);
    gemm_small(ctx_h, decT, dec_b, qbuf, nullptr, Bn, Dd, 384, 0);
    dec_step_score<<<Bn, 128, 0, stream>>>(qbuf, x_f32, mask, cur);
  }
  dec_final<<<cdiv(Bn, 256), 256, 0, stream>>>(cur, tour_out);
  tour_length<<<cdiv(Bn, 256), 256, 0, stream>>>(coords, tour_out, len_out);
  copy_enc<<<cdiv((size_t)TOK * Dd, 256), 256, 0, stream>>>(x_f32, enc_out);
}
